// ConfigurableGATEncoder_13159779795150
// MI455X (gfx1250) — matrix operand order repaired and checked by reading, not compiled, not run on hardware
//
#include <hip/hip_runtime.h>
#include <hip/hip_bf16.h>

#define N_NODES 50000
#define E_EDGES 800000
#define ETOT    (E_EDGES + N_NODES)
#define NEG_SLOPE 0.2f
#define BN_EPS 1e-5f

typedef __attribute__((ext_vector_type(16))) __bf16 v16bf;
typedef __attribute__((ext_vector_type(8)))  float  v8f;

struct __attribute__((aligned(16))) U4 { unsigned int x[4]; };
union Frag { v16bf v; U4 q[2]; unsigned short u[16]; };

// ---------- helpers ----------
__device__ __forceinline__ unsigned short f2bf(float f) {
    unsigned u = __float_as_uint(f);
    unsigned r = (u + 0x7FFFu + ((u >> 16) & 1u)) >> 16;
    return (unsigned short)r;
}
// order-preserving float <-> uint encoding for atomicMax
__device__ __forceinline__ unsigned fenc(float f) {
    unsigned u = __float_as_uint(f);
    return (u & 0x80000000u) ? ~u : (u | 0x80000000u);
}
__device__ __forceinline__ float fdec(unsigned u) {
    return (u & 0x80000000u) ? __uint_as_float(u & 0x7FFFFFFFu) : __uint_as_float(~u);
}
__device__ __forceinline__ void edge_sd(const int* __restrict__ ei, int e, int& s, int& d) {
    if (e < E_EDGES) { s = ei[e]; d = ei[E_EDGES + e]; }
    else             { s = e - E_EDGES; d = s; }
}
__device__ __forceinline__ float lrelu(float v) { return v > 0.f ? v : NEG_SLOPE * v; }

// ---------- input embedding: h0 = [x[:, :16], cell_emb[id]] -> bf16, padded 48->64 ----------
__global__ void embed_kernel(const float* __restrict__ x, const float* __restrict__ cell_emb,
                             unsigned short* __restrict__ hbf) {
    int n = blockIdx.x;          // one block per node
    int t = threadIdx.x;         // 64 threads (ldk = 64)
    const float* row = x + (size_t)n * 17;
    float v = 0.f;
    if (t < 16)       v = row[t];
    else if (t < 48)  { int id = (int)row[16]; v = cell_emb[(size_t)id * 32 + (t - 16)]; }
    hbf[(size_t)n * 64 + t] = f2bf(v);
}

// ---------- repack W (fp32 [Kin,Kout]) into per-lane WMMA B fragments (bf16) ----------
// layout: Bpack[((ct*KB + kb)*32 + lane)*16 + j], j -> K = kb*32 + 16*(lane>>4) + j, N = ct*16 + (lane&15)
__global__ void packB_kernel(const float* __restrict__ W, unsigned short* __restrict__ Bpack,
                             int Kin, int KB, int ColT, int Kout) {
    int idx = blockIdx.x * blockDim.x + threadIdx.x;
    int total = ColT * KB * 32;
    if (idx >= total) return;
    int lane = idx & 31;
    int kb   = (idx >> 5) % KB;
    int ct   = idx / (KB * 32);
    int n = lane & 15, khalf = lane >> 4;
    unsigned short* dst = Bpack + (size_t)idx * 16;
#pragma unroll
    for (int j = 0; j < 16; ++j) {
        int k = kb * 32 + khalf * 16 + j;
        float v = (k < Kin) ? W[(size_t)k * Kout + ct * 16 + n] : 0.f;
        dst[j] = f2bf(v);
    }
}

// ---------- bf16 WMMA GEMM: C[N,Kout] = A[N,ldk(bf16)] * Bpack ----------
// one wave -> one 16x16 output tile; K-loop in steps of 32 with v_wmma_f32_16x16x32_bf16
__global__ void gemm_wmma_kernel(const unsigned short* __restrict__ Abf,
                                 const unsigned short* __restrict__ Bpack,
                                 float* __restrict__ C,
                                 int ldk, int KB, int ColT, int Kout) {
    int wave = (blockIdx.x * blockDim.x + threadIdx.x) >> 5;
    int lane = threadIdx.x & 31;
    int total = (N_NODES / 16) * ColT;
    if (wave >= total) return;
    int ct = wave % ColT;
    int rt = wave / ColT;

    int m  = lane & 15;
    int hi = lane >> 4;
    int ka0 = hi ? 16 : 0;     // A fragment chunk 0 K-offset (contiguous k = 16*hi + s, matches Bpack)
    int ka1 = hi ? 24 : 8;     // A fragment chunk 1 K-offset
    const unsigned short* arow = Abf + (size_t)(rt * 16 + m) * ldk;
    const unsigned short* bp   = Bpack + (size_t)(ct * KB) * 512 + (size_t)lane * 16;

    v8f acc = {};
    for (int kb = 0; kb < KB; ++kb) {
        Frag a, b;
        a.q[0] = *(const U4*)(arow + kb * 32 + ka0);
        a.q[1] = *(const U4*)(arow + kb * 32 + ka1);
        b.q[0] = *(const U4*)(bp + (size_t)kb * 512);
        b.q[1] = *(const U4*)(bp + (size_t)kb * 512 + 8);
        acc = __builtin_amdgcn_wmma_f32_16x16x32_bf16(false, a.v, false, b.v,
                                                      (short)0, acc, false, false);
    }
    // C/D layout: lanes 0-15 rows M=0..7, lanes 16-31 rows M=8..15; col = lane&15
    int col   = ct * 16 + (lane & 15);
    int rbase = rt * 16 + hi * 8;
#pragma unroll
    for (int r = 0; r < 8; ++r)
        C[(size_t)(rbase + r) * Kout + col] = acc[r];
}

// ---------- attention projections: al_s/al_d[n,h] = sum_c xl[n,h,c]*a[h,c] ----------
__global__ void attn_coef_kernel(const float* __restrict__ xl,
                                 const float* __restrict__ a_s, const float* __restrict__ a_d,
                                 float* __restrict__ als, float* __restrict__ ald,
                                 int H, int C) {
    int i = blockIdx.x * blockDim.x + threadIdx.x;
    if (i >= N_NODES * H) return;
    int n = i / H, h = i % H;
    const float* xr = xl + (size_t)n * H * C + (size_t)h * C;
    float ss = 0.f, sd = 0.f;
    for (int c = 0; c < C; ++c) {
        float v = xr[c];
        ss = fmaf(v, a_s[h * C + c], ss);
        sd = fmaf(v, a_d[h * C + c], sd);
    }
    als[i] = ss; ald[i] = sd;
}

// ---------- edge pass 1: segment max of leaky_relu(als[src]+ald[dst]) ----------
__global__ void edge_max_kernel(const int* __restrict__ ei,
                                const float* __restrict__ als, const float* __restrict__ ald,
                                unsigned* __restrict__ emax, int H) {
    int e = blockIdx.x * blockDim.x + threadIdx.x;
    if (e >= ETOT) return;
    int s, d; edge_sd(ei, e, s, d);
    for (int h = 0; h < H; ++h) {
        float v = lrelu(als[s * H + h] + ald[d * H + h]);
        atomicMax(&emax[d * H + h], fenc(v));
    }
}

// ---------- edge pass 2: p = exp(e - max), denom = segment_sum(p) ----------
__global__ void edge_exp_kernel(const int* __restrict__ ei,
                                const float* __restrict__ als, const float* __restrict__ ald,
                                const unsigned* __restrict__ emax,
                                float* __restrict__ p, float* __restrict__ denom, int H) {
    int e = blockIdx.x * blockDim.x + threadIdx.x;
    if (e >= ETOT) return;
    int s, d; edge_sd(ei, e, s, d);
    for (int h = 0; h < H; ++h) {
        float v  = lrelu(als[s * H + h] + ald[d * H + h]);
        float pv = __expf(v - fdec(emax[d * H + h]));
        p[(size_t)e * H + h] = pv;
        atomicAdd(&denom[d * H + h], pv);
    }
}

// ---------- edge pass 3: out[dst] += xl[src] * alpha ----------
__global__ void edge_scatter_kernel(const int* __restrict__ ei, const float* __restrict__ xl,
                                    const float* __restrict__ p, const float* __restrict__ denom,
                                    float* __restrict__ out, int H, int C) {
    int e = blockIdx.x;          // one block per edge, blockDim = H*C
    int t = threadIdx.x;
    int s, d; edge_sd(ei, e, s, d);
    const float* src_row = xl + (size_t)s * H * C;
    __builtin_prefetch(src_row, 0, 0);      // global_prefetch_b8
    int h = t / C;
    float alpha = p[(size_t)e * H + h] / (denom[d * H + h] + 1e-16f);
    atomicAdd(&out[(size_t)d * H * C + t], src_row[t] * alpha);
}

// ---------- BN stats: per-channel sum / sumsq (LDS tree reduction) ----------
__global__ void bn_stats_kernel(const float* __restrict__ out,
                                float* __restrict__ sum, float* __restrict__ sumsq, int HC) {
    __shared__ float s1[256], s2[256];
    int ch = blockIdx.x;
    float a = 0.f, b = 0.f;
    for (int n = blockIdx.y * blockDim.x + threadIdx.x; n < N_NODES; n += gridDim.y * blockDim.x) {
        float v = out[(size_t)n * HC + ch];
        a += v; b = fmaf(v, v, b);
    }
    s1[threadIdx.x] = a; s2[threadIdx.x] = b;
    __syncthreads();
    for (int o = 128; o > 0; o >>= 1) {
        if (threadIdx.x < o) { s1[threadIdx.x] += s1[threadIdx.x + o]; s2[threadIdx.x] += s2[threadIdx.x + o]; }
        __syncthreads();
    }
    if (threadIdx.x == 0) { atomicAdd(&sum[ch], s1[0]); atomicAdd(&sumsq[ch], s2[0]); }
}

// ---------- BN + ELU, write next layer's bf16 input (bias cancels under BN mean-subtract) ----------
__global__ void bn_apply_kernel(const float* __restrict__ out,
                                const float* __restrict__ sum, const float* __restrict__ sumsq,
                                const float* __restrict__ g, const float* __restrict__ beta,
                                unsigned short* __restrict__ hbf_next, int HC) {
    int i = blockIdx.x * blockDim.x + threadIdx.x;
    if (i >= N_NODES * HC) return;
    int ch = i % HC;
    const float invN = 1.f / (float)N_NODES;
    float m   = sum[ch] * invN;
    float var = sumsq[ch] * invN - m * m;
    float y = (out[i] - m) * __frsqrt_rn(var + BN_EPS) * g[ch] + beta[ch];
    y = y > 0.f ? y : (__expf(y) - 1.f);   // ELU
    hbf_next[i] = f2bf(y);                 // ldk_next == HC (256 or 128), no pad needed
}

// ---------- final: y = out2 + b2 ----------
__global__ void final_bias_kernel(const float* __restrict__ out, const float* __restrict__ b,
                                  float* __restrict__ y, int HC) {
    int i = blockIdx.x * blockDim.x + threadIdx.x;
    if (i >= N_NODES * HC) return;
    y[i] = out[i] + b[i % HC];
}

// =====================================================================
extern "C" void kernel_launch(void* const* d_in, const int* in_sizes, int n_in,
                              void* d_out, int out_size, void* d_ws, size_t ws_size,
                              hipStream_t stream) {
    const float* x        = (const float*)d_in[0];
    const int*   ei       = (const int*)  d_in[1];
    const float* cell_emb = (const float*)d_in[2];
    const float* W[3]     = { (const float*)d_in[3],  (const float*)d_in[7],  (const float*)d_in[11] };
    const float* a_s[3]   = { (const float*)d_in[4],  (const float*)d_in[8],  (const float*)d_in[12] };
    const float* a_d[3]   = { (const float*)d_in[5],  (const float*)d_in[9],  (const float*)d_in[13] };
    const float* bias2    =   (const float*)d_in[14];
    const float* g[2]     = { (const float*)d_in[15], (const float*)d_in[17] };
    const float* beta[2]  = { (const float*)d_in[16], (const float*)d_in[18] };
    float* y_out = (float*)d_out;

    // layer configs
    const int Kin[3]  = { 48, 256, 128 };
    const int KinP[3] = { 64, 256, 128 };     // padded to multiple of 32
    const int Hh[3]   = { 4, 2, 1 };
    const int Cc[3]   = { 64, 64, 32 };

    // ---- workspace partition (256B aligned) ----
    char* w = (char*)d_ws;
    auto alloc = [&](size_t bytes) { char* r = w; w += (bytes + 255) & ~(size_t)255; return r; };
    unsigned short* hbfA = (unsigned short*)alloc((size_t)N_NODES * 256 * 2);
    unsigned short* hbfB = (unsigned short*)alloc((size_t)N_NODES * 256 * 2);
    float*    xl    = (float*)   alloc((size_t)N_NODES * 256 * 4);
    float*    outb  = (float*)   alloc((size_t)N_NODES * 256 * 4);
    float*    als   = (float*)   alloc((size_t)N_NODES * 4 * 4);
    float*    ald   = (float*)   alloc((size_t)N_NODES * 4 * 4);
    unsigned* emax  = (unsigned*)alloc((size_t)N_NODES * 4 * 4);
    float*    denom = (float*)   alloc((size_t)N_NODES * 4 * 4);
    float*    pbuf  = (float*)   alloc((size_t)ETOT * 4 * 4);
    unsigned short* Bpack = (unsigned short*)alloc(128 * 1024);
    float*    sum   = (float*)   alloc(256 * 4);
    float*    sumsq = (float*)   alloc(256 * 4);
    (void)ws_size; (void)n_in; (void)in_sizes; (void)out_size;

    // ---- embed: x + cell_emb -> hbfA [N,64] bf16 (zero-padded 48..63) ----
    embed_kernel<<<N_NODES, 64, 0, stream>>>(x, cell_emb, hbfA);

    unsigned short* hin = hbfA;
    unsigned short* hnext = hbfB;

    for (int li = 0; li < 3; ++li) {
        const int H = Hh[li], C = Cc[li], HC = H * C;
        const int KB = KinP[li] / 32, ColT = HC / 16;

        // pack weights into WMMA B fragments
        {
            int total = ColT * KB * 32;
            packB_kernel<<<(total + 255) / 256, 256, 0, stream>>>(W[li], Bpack, Kin[li], KB, ColT, HC);
        }
        // GEMM: xl = h @ W   (bf16 WMMA, f32 accumulate)
        {
            int tiles = (N_NODES / 16) * ColT;
            int blocks = (tiles * 32 + 255) / 256;
            gemm_wmma_kernel<<<blocks, 256, 0, stream>>>(hin, Bpack, xl, KinP[li], KB, ColT, HC);
        }
        // attention projections
        attn_coef_kernel<<<(N_NODES * H + 255) / 256, 256, 0, stream>>>(xl, a_s[li], a_d[li], als, ald, H, C);

        // per-call zeroing (graph-capture-safe)
        hipMemsetAsync(emax,  0, (size_t)N_NODES * H * 4, stream);
        hipMemsetAsync(denom, 0, (size_t)N_NODES * H * 4, stream);
        hipMemsetAsync(outb,  0, (size_t)N_NODES * HC * 4, stream);

        // segment softmax + message scatter
        edge_max_kernel    <<<(ETOT + 255) / 256, 256, 0, stream>>>(ei, als, ald, emax, H);
        edge_exp_kernel    <<<(ETOT + 255) / 256, 256, 0, stream>>>(ei, als, ald, emax, pbuf, denom, H);
        edge_scatter_kernel<<<ETOT, HC, 0, stream>>>(ei, xl, pbuf, denom, outb, H, C);

        if (li < 2) {
            hipMemsetAsync(sum,   0, HC * 4, stream);
            hipMemsetAsync(sumsq, 0, HC * 4, stream);
            bn_stats_kernel<<<dim3(HC, 32), 256, 0, stream>>>(outb, sum, sumsq, HC);
            // BN+ELU -> next layer's bf16 input (layer bias cancels under mean-subtraction)
            bn_apply_kernel<<<(N_NODES * HC + 255) / 256, 256, 0, stream>>>(
                outb, sum, sumsq, g[li], beta[li], hnext, HC);
            unsigned short* t = hin; hin = hnext; hnext = t;
        } else {
            final_bias_kernel<<<(N_NODES * HC + 255) / 256, 256, 0, stream>>>(outb, bias2, y_out, HC);
        }
    }
}
